// MultiHeadLatentAttention_82386062672303
// MI455X (gfx1250) — matrix operand order repaired and checked by reading, not compiled, not run on hardware
//
#include <hip/hip_runtime.h>
#include <hip/hip_bf16.h>

typedef __bf16 bf16_t;
typedef __attribute__((ext_vector_type(16))) __bf16 bf16x16;
typedef __attribute__((ext_vector_type(8)))  __bf16 bf16x8;
typedef __attribute__((ext_vector_type(8)))  float  f32x8;

// ---------------------------------------------------------------------------
// WMMA helper: D = A(16x32 bf16) * B(32x16 bf16) + C(16x16 f32)
// ---------------------------------------------------------------------------
__device__ __forceinline__ f32x8 wmma_bf16(bf16x16 a, bf16x16 b, f32x8 c) {
  return __builtin_amdgcn_wmma_f32_16x16x32_bf16(
      /*neg_a=*/false, a, /*neg_b=*/false, b,
      /*c_mod=*/(short)0, c, /*reuse_a=*/false, /*reuse_b=*/false);
}

// A-fragment (16x32, bf16): lane l holds row (l&15); K = (l>>4)*16 + {0..15}
// (contiguous, matching B).  Two contiguous 16B loads.
__device__ __forceinline__ bf16x16 load_a_frag(const bf16_t* p, int ld) {
  int lane = threadIdx.x & 31;
  const bf16_t* q = p + (size_t)(lane & 15) * ld + ((lane >> 4) * 16);
  union { bf16x16 v; bf16x8 h[2]; } u;
  u.h[0] = *(const bf16x8*)(q);
  u.h[1] = *(const bf16x8*)(q + 8);
  return u.v;
}

// B-fragment (32x16, bf16) sourced from a row-major [N,K] matrix: lane l holds
// column (l&15); K = (l>>4)*16 + {0..15} -> one contiguous 32B load.
__device__ __forceinline__ bf16x16 load_b_frag(const bf16_t* __restrict__ p, int ld) {
  int lane = threadIdx.x & 31;
  return *(const bf16x16*)(p + (size_t)(lane & 15) * ld + ((lane >> 4) * 16));
}

// Async 16B global -> LDS copy (per-lane), tracked by ASYNCcnt.
__device__ __forceinline__ void async_copy_16B(unsigned lds_off, const bf16_t* g) {
  asm volatile("global_load_async_to_lds_b128 %0, %1, off"
               :: "v"(lds_off), "v"(g) : "memory");
}
__device__ __forceinline__ void wait_async0() {
  asm volatile("s_wait_asynccnt 0" ::: "memory");
}

// ---------------------------------------------------------------------------
// f32 -> bf16 convert
// ---------------------------------------------------------------------------
__global__ void f32_to_bf16_kernel(const float* __restrict__ in, bf16_t* __restrict__ out,
                                   size_t n) {
  size_t i = (size_t)blockIdx.x * blockDim.x + threadIdx.x;
  if (i < n) out[i] = (bf16_t)in[i];
}

// ---------------------------------------------------------------------------
// RMS norm over last dim, fp32 in -> bf16 out
// ---------------------------------------------------------------------------
__global__ void rmsnorm_bf16_kernel(const float* __restrict__ in, const float* __restrict__ w,
                                    bf16_t* __restrict__ out, int D) {
  __shared__ float red[8];
  size_t row = blockIdx.x;
  const float* x = in + row * (size_t)D;
  bf16_t* o = out + row * (size_t)D;
  float s = 0.f;
  for (int i = threadIdx.x; i < D; i += blockDim.x) { float v = x[i]; s += v * v; }
  for (int off = 16; off >= 1; off >>= 1) s += __shfl_xor(s, off, 32);
  if ((threadIdx.x & 31) == 0) red[threadIdx.x >> 5] = s;
  __syncthreads();
  float tot = 0.f;
#pragma unroll
  for (int i = 0; i < 8; ++i) tot += red[i];
  float rinv = rsqrtf(tot / (float)D + 1.1920929e-07f);
  for (int i = threadIdx.x; i < D; i += blockDim.x)
    o[i] = (bf16_t)(x[i] * rinv * w[i]);
}

// ---------------------------------------------------------------------------
// GEMM: C[M,N] = A[M,K](bf16,rm) * W[N,K](bf16,rm)^T + bias[N], C fp32 rm.
// Block tile 64(M) x 128(N), 4 waves; each wave computes 64x32.
// A tile (64x32) is staged in LDS via GLOBAL_LOAD_ASYNC_TO_LDS_B128 with
// double buffering (ASYNCcnt); B frags load contiguously from global.
// LDS row pitch = 40 elems (80B): bank-conflict-free A-fragment ds reads.
// ---------------------------------------------------------------------------
constexpr int GA_PITCH = 40;

__global__ void gemm_bf16_kernel(const bf16_t* __restrict__ A, const bf16_t* __restrict__ W,
                                 const float* __restrict__ bias, float* __restrict__ C,
                                 int M, int N, int K) {
  __shared__ bf16_t tileA[2][64 * GA_PITCH];

  int wave = threadIdx.x >> 5;
  int n0 = blockIdx.x * 128 + wave * 32;
  int m0 = blockIdx.y * 64;

  const bf16_t* Wp0 = W + (size_t)n0 * K;
  const bf16_t* Wp1 = W + (size_t)(n0 + 16) * K;

  f32x8 acc[4][2];
#pragma unroll
  for (int im = 0; im < 4; ++im) {
    acc[im][0] = {};
    acc[im][1] = {};
  }

  // stage one 64x32 A tile: 256 16B units, 128 threads -> 2 async issues each
  auto issue_tile = [&](int kt, int buf) {
#pragma unroll
    for (int t = 0; t < 2; ++t) {
      int u = (int)threadIdx.x + t * 128;
      int row = u >> 2, q = u & 3;
      const bf16_t* g = A + (size_t)(m0 + row) * K + kt * 32 + q * 8;
      unsigned lds = (unsigned)(unsigned long long)&tileA[buf][row * GA_PITCH + q * 8];
      async_copy_16B(lds, g);
    }
  };

  int nk = K / 32;
  issue_tile(0, 0);
  int cur = 0;

  for (int kt = 0; kt < nk; ++kt) {
    wait_async0();        // own wave's async writes to tileA[cur] complete
    __syncthreads();      // publish tile; also proves buf (1-cur) fully consumed

    if (kt + 1 < nk) issue_tile(kt + 1, 1 - cur);

    int k0 = kt * 32;
    bf16x16 b0 = load_b_frag(Wp0 + k0, K);
    bf16x16 b1 = load_b_frag(Wp1 + k0, K);
#pragma unroll
    for (int im = 0; im < 4; ++im) {
      bf16x16 a = load_a_frag(&tileA[cur][im * 16 * GA_PITCH], GA_PITCH);
      acc[im][0] = wmma_bf16(a, b0, acc[im][0]);
      acc[im][1] = wmma_bf16(a, b1, acc[im][1]);
    }
    cur ^= 1;
  }

  int lane = threadIdx.x & 31;
  int cm = (lane >> 4) * 8, cn = lane & 15;
  float bv0 = bias[n0 + cn];
  float bv1 = bias[n0 + 16 + cn];
#pragma unroll
  for (int im = 0; im < 4; ++im) {
#pragma unroll
    for (int r = 0; r < 8; ++r) {
      size_t row = (size_t)(m0 + im * 16 + cm + r);
      C[row * N + n0 + cn]      = acc[im][0][r] + bv0;
      C[row * N + n0 + 16 + cn] = acc[im][1][r] + bv1;
    }
  }
}

// ---------------------------------------------------------------------------
// Reorder / concat:  qf[BS,2048], kvf[BS,3072], krf[BS,1024] (fp32)
//   -> Q  bf16 [B,H,S,128]   (q_nope||q_rope already contiguous per head)
//   -> K  bf16 [B,H,S,128]   (kv[...,:64] || k_rope per head)
//   -> Vt bf16 [B,H,128,S]   (transposed V = kv[...,64:192])
// ---------------------------------------------------------------------------
__global__ void interleave_kernel(const float* __restrict__ qf, const float* __restrict__ kvf,
                                  const float* __restrict__ krf, bf16_t* __restrict__ Qb,
                                  bf16_t* __restrict__ Kb, bf16_t* __restrict__ Vt,
                                  int H, int S) {
  int d  = threadIdx.x;   // 0..127
  int s  = blockIdx.x;    // 0..S-1
  int bh = blockIdx.y;    // 0..B*H-1
  int b = bh / H, h = bh % H;
  size_t rowm = (size_t)b * S + s;
  float qv = qf[rowm * 2048 + h * 128 + d];
  float kv = (d < 64) ? kvf[rowm * 3072 + h * 192 + d]
                      : krf[rowm * 1024 + h * 64 + (d - 64)];
  float vv = kvf[rowm * 3072 + h * 192 + 64 + d];
  size_t o = ((size_t)bh * S + s) * 128 + d;
  Qb[o] = (bf16_t)qv;
  Kb[o] = (bf16_t)kv;
  Vt[((size_t)bh * 128 + d) * S + s] = (bf16_t)vv;
}

// ---------------------------------------------------------------------------
// Causal flash attention. Q,K: [B,H,S,128] bf16; Vt: [B,H,128,S] bf16.
// One wave owns 16 query rows; 4 independent waves per block.
// Per 32-key tile: 8 WMMAs (QK^T) + online softmax + 8 WMMAs (PV).
// Output: attn bf16 [B,S,H*128].
// ---------------------------------------------------------------------------
constexpr int AT_WAVES = 4;
constexpr int P_PITCH  = 40;   // 80B row pitch: conflict-free & 16B aligned

__global__ void mla_attention_kernel(const bf16_t* __restrict__ Q, const bf16_t* __restrict__ K,
                                     const bf16_t* __restrict__ Vt, bf16_t* __restrict__ Aout,
                                     int H, int S) {
  __shared__ bf16_t ldsP[AT_WAVES][16 * P_PITCH];
  const int DKH = 128;
  int wave = threadIdx.x >> 5, lane = threadIdx.x & 31;
  int bh = blockIdx.x;
  int b = bh / H, h = bh % H;
  int q0 = (blockIdx.y * AT_WAVES + wave) * 16;

  const bf16_t* Qh = Q + (size_t)bh * S * DKH;
  const bf16_t* Kh = K + (size_t)bh * S * DKH;
  const bf16_t* Vh = Vt + (size_t)bh * DKH * S;

  // Q fragments for this 16-row tile (reused across all key tiles)
  bf16x16 qfr[4];
#pragma unroll
  for (int c = 0; c < 4; ++c)
    qfr[c] = load_a_frag(Qh + (size_t)q0 * DKH + c * 32, DKH);

  f32x8 accO[8];
#pragma unroll
  for (int c = 0; c < 8; ++c) accO[c] = {};
  float m[8], l[8];
#pragma unroll
  for (int r = 0; r < 8; ++r) { m[r] = -1e30f; l[r] = 0.f; }

  const float scale = 0.08838834764831845f;  // 1/sqrt(128)
  int cm = (lane >> 4) * 8, cn = lane & 15;
  bf16_t* myP = ldsP[wave];

  for (int kb = 0; kb < q0 + 16; kb += 32) {
    // ---- scores S = Q * K_tile^T  (16 x 32) ----
    f32x8 s0 = {}, s1 = {};
#pragma unroll
    for (int c = 0; c < 4; ++c) {
      bf16x16 k0f = load_b_frag(Kh + (size_t)kb * DKH + c * 32, DKH);
      bf16x16 k1f = load_b_frag(Kh + (size_t)(kb + 16) * DKH + c * 32, DKH);
      s0 = wmma_bf16(qfr[c], k0f, s0);
      s1 = wmma_bf16(qfr[c], k1f, s1);
    }

    // ---- scale + causal mask ----
#pragma unroll
    for (int r = 0; r < 8; ++r) {
      int qrow = q0 + cm + r;
      float v0 = s0[r] * scale;
      float v1 = s1[r] * scale;
      if (kb + cn > qrow)      v0 = -1e30f;
      if (kb + 16 + cn > qrow) v1 = -1e30f;
      s0[r] = v0; s1[r] = v1;
    }

    // ---- online softmax (row stats are uniform across each 16-lane half) ----
    float alpha[8];
#pragma unroll
    for (int r = 0; r < 8; ++r) {
      float rm = fmaxf(s0[r], s1[r]);
      for (int o = 8; o >= 1; o >>= 1) rm = fmaxf(rm, __shfl_xor(rm, o, 32));
      float mn = fmaxf(m[r], rm);
      alpha[r] = __expf(m[r] - mn);
      float e0 = __expf(s0[r] - mn);
      float e1 = __expf(s1[r] - mn);
      float rs = e0 + e1;
      for (int o = 8; o >= 1; o >>= 1) rs += __shfl_xor(rs, o, 32);
      l[r] = l[r] * alpha[r] + rs;
      m[r] = mn;
      s0[r] = e0; s1[r] = e1;
    }
#pragma unroll
    for (int c = 0; c < 8; ++c)
#pragma unroll
      for (int r = 0; r < 8; ++r) accO[c][r] *= alpha[r];

    // ---- P (16x32) -> bf16 via LDS, re-read in A-fragment layout ----
#pragma unroll
    for (int r = 0; r < 8; ++r) {
      myP[(cm + r) * P_PITCH + cn]      = (bf16_t)s0[r];
      myP[(cm + r) * P_PITCH + 16 + cn] = (bf16_t)s1[r];
    }
    asm volatile("s_wait_dscnt 0" ::: "memory");
    bf16x16 pf = load_a_frag(myP, P_PITCH);

    // ---- O += P * V_tile  (V^T gives contiguous B-fragment rows) ----
#pragma unroll
    for (int c = 0; c < 8; ++c) {
      bf16x16 vf = load_b_frag(Vh + (size_t)(c * 16) * S + kb, S);
      accO[c] = wmma_bf16(pf, vf, accO[c]);
    }
  }

  // ---- normalize and store to attn [B,S,H*128] ----
#pragma unroll
  for (int r = 0; r < 8; ++r) {
    float inv = 1.0f / l[r];
    int row = q0 + cm + r;
    size_t base = ((size_t)b * S + row) * 2048 + (size_t)h * 128;
#pragma unroll
    for (int c = 0; c < 8; ++c)
      Aout[base + c * 16 + cn] = (bf16_t)(accO[c][r] * inv);
  }
}

extern "C" void kernel_launch(void* const* d_in, const int* in_sizes, int n_in,
                              void* d_out, int out_size, void* d_ws, size_t ws_size,
                              hipStream_t stream) {
  const float* x   = (const float*)d_in[0];
  const float* qdw = (const float*)d_in[1];
  const float* qdb = (const float*)d_in[2];
  const float* qnw = (const float*)d_in[3];
  const float* quw = (const float*)d_in[4];
  const float* qub = (const float*)d_in[5];
  const float* kdw = (const float*)d_in[6];
  const float* kdb = (const float*)d_in[7];
  const float* knw = (const float*)d_in[8];
  const float* kuw = (const float*)d_in[9];
  const float* kub = (const float*)d_in[10];
  const float* krw = (const float*)d_in[11];
  const float* krb = (const float*)d_in[12];
  const float* ow  = (const float*)d_in[13];
  const float* ob  = (const float*)d_in[14];
  float* out = (float*)d_out;
  (void)in_sizes; (void)n_in; (void)out_size; (void)ws_size;

  constexpr int Bn = 2, S = 2048, DM = 2048, H = 16, DK = 128, DC = 512, DCQ = 1024, DR = 64;
  constexpr int BS = Bn * S;

  size_t off = 0;
  char* base = (char*)d_ws;
  auto alloc = [&](size_t nbytes) -> char* {
    char* p = base + off;
    off += (nbytes + 255) & ~(size_t)255;
    return p;
  };

  bf16_t* xb   = (bf16_t*)alloc((size_t)BS * DM * 2);
  bf16_t* wqd  = (bf16_t*)alloc((size_t)DCQ * DM * 2);
  bf16_t* wqu  = (bf16_t*)alloc((size_t)(H * DK) * DCQ * 2);
  bf16_t* wkd  = (bf16_t*)alloc((size_t)DC * DM * 2);
  bf16_t* wku  = (bf16_t*)alloc((size_t)(H * 192) * DC * 2);
  bf16_t* wkr  = (bf16_t*)alloc((size_t)(H * DR) * DM * 2);
  bf16_t* wo   = (bf16_t*)alloc((size_t)DM * DM * 2);
  float*  cqf  = (float*) alloc((size_t)BS * DCQ * 4);
  bf16_t* cqb  = (bf16_t*)alloc((size_t)BS * DCQ * 2);
  float*  ckvf = (float*) alloc((size_t)BS * DC * 4);
  bf16_t* ckvb = (bf16_t*)alloc((size_t)BS * DC * 2);
  float*  qfb  = (float*) alloc((size_t)BS * (H * DK) * 4);
  float*  kvf  = (float*) alloc((size_t)BS * (H * 192) * 4);
  float*  krf  = (float*) alloc((size_t)BS * (H * DR) * 4);
  bf16_t* Qb   = (bf16_t*)alloc((size_t)Bn * H * S * DK * 2);
  bf16_t* Kb   = (bf16_t*)alloc((size_t)Bn * H * S * DK * 2);
  bf16_t* Vt   = (bf16_t*)alloc((size_t)Bn * H * DK * S * 2);
  bf16_t* atb  = (bf16_t*)alloc((size_t)BS * DM * 2);

  auto cvt = [&](const float* src, bf16_t* dst, size_t n) {
    f32_to_bf16_kernel<<<(unsigned)((n + 255) / 256), 256, 0, stream>>>(src, dst, n);
  };
  cvt(x,   xb,  (size_t)BS * DM);
  cvt(qdw, wqd, (size_t)DCQ * DM);
  cvt(quw, wqu, (size_t)(H * DK) * DCQ);
  cvt(kdw, wkd, (size_t)DC * DM);
  cvt(kuw, wku, (size_t)(H * 192) * DC);
  cvt(krw, wkr, (size_t)(H * DR) * DM);
  cvt(ow,  wo,  (size_t)DM * DM);

  auto gemm = [&](const bf16_t* A, const bf16_t* W, const float* bias, float* C,
                  int M, int N, int Kd) {
    dim3 g(N / 128, M / 64);
    gemm_bf16_kernel<<<g, 128, 0, stream>>>(A, W, bias, C, M, N, Kd);
  };

  // down projections + rope keys from x
  gemm(xb, wqd, qdb, cqf,  BS, DCQ,    DM);
  gemm(xb, wkd, kdb, ckvf, BS, DC,     DM);
  gemm(xb, wkr, krb, krf,  BS, H * DR, DM);

  // RMS norms
  rmsnorm_bf16_kernel<<<BS, 256, 0, stream>>>(cqf,  qnw, cqb,  DCQ);
  rmsnorm_bf16_kernel<<<BS, 256, 0, stream>>>(ckvf, knw, ckvb, DC);

  // up projections
  gemm(cqb,  wqu, qub, qfb, BS, H * DK,  DCQ);
  gemm(ckvb, wku, kub, kvf, BS, H * 192, DC);

  // reorder to [B,H,S,128] Q/K and [B,H,128,S] V^T (bf16)
  interleave_kernel<<<dim3(S, Bn * H), 128, 0, stream>>>(qfb, kvf, krf, Qb, Kb, Vt, H, S);

  // causal flash attention -> attn [B,S,2048] bf16
  mla_attention_kernel<<<dim3(Bn * H, S / (16 * AT_WAVES)), 32 * AT_WAVES, 0, stream>>>(
      Qb, Kb, Vt, atb, H, S);

  // output projection -> fp32 d_out
  gemm(atb, wo, ob, out, BS, DM, DM);
}
